// GraphAttentionLayer_50972671869364
// MI455X (gfx1250) — matrix operand order repaired and checked by reading, not compiled, not run on hardware
//
#include <hip/hip_runtime.h>

#define GAT_N     8192
#define GAT_INF   128
#define GAT_NH    64
#define GAT_OUTF  128
#define GAT_KS    4                      // key-dimension split -> 2048 waves
#define GAT_QT    (GAT_N / 16)           // 512 query tiles
#define GAT_LOG2E 1.44269504088896f

typedef __attribute__((ext_vector_type(16))) _Float16 v16h;
typedef __attribute__((ext_vector_type(8)))  _Float16 v8h;
typedef __attribute__((ext_vector_type(8)))  float    v8f;
typedef __attribute__((ext_vector_type(2)))  int      v2i;

// ---------------------------------------------------------------------------
// Kernel 1: projections.  Whs = (h@Ws)*log2(e)  (row-major f16, [N][64]) so
// score WMMAs produce log2-domain scores and exp is a bare v_exp_f32.
// Wht = h@Wt (row-major f16, [N][64]); WhcT = (h@Wc)^T (f16, [128][N]) so the
// PV B-operand is a contiguous 16-half load per lane.
// ---------------------------------------------------------------------------
__global__ __launch_bounds__(256) void gat_proj_kernel(
    const float* __restrict__ h,
    const float* __restrict__ Ws,
    const float* __restrict__ Wt,
    const float* __restrict__ Wc,
    _Float16* __restrict__ Whs,
    _Float16* __restrict__ Wht,
    _Float16* __restrict__ WhcT)
{
  const int row = blockIdx.x;
  const int col = threadIdx.x;          // 0..255: [0,64)=Ws, [64,128)=Wt, [128,256)=Wc
  const float* __restrict__ hr = h + row * GAT_INF;

  float acc = 0.0f;
  if (col < GAT_NH) {
#pragma unroll 8
    for (int k = 0; k < GAT_INF; ++k) acc = fmaf(hr[k], Ws[k * GAT_NH + col], acc);
    Whs[row * GAT_NH + col] = (_Float16)(acc * GAT_LOG2E);
  } else if (col < 2 * GAT_NH) {
    const int c = col - GAT_NH;
#pragma unroll 8
    for (int k = 0; k < GAT_INF; ++k) acc = fmaf(hr[k], Wt[k * GAT_NH + c], acc);
    Wht[row * GAT_NH + c] = (_Float16)acc;
  } else {
    const int c = col - 2 * GAT_NH;
#pragma unroll 8
    for (int k = 0; k < GAT_INF; ++k) acc = fmaf(hr[k], Wc[k * GAT_OUTF + c], acc);
    WhcT[c * GAT_N + row] = (_Float16)acc;
  }
}

// ---------------------------------------------------------------------------
// Kernel 2: fused masked attention, split-K over keys.
// Block (qt, ks): one wave32 owning 16 query rows, streaming keys
// [ks*2048, (ks+1)*2048) in tiles of 32.  Score tile0 covers the EVEN keys
// (j0+2c), tile1 the ODD keys (j0+2c+1) -- chosen purely by which Wht rows
// feed the B-operand.  Consequences:
//   * each lane's two adj mask values are memory-adjacent -> 8x b64 NT loads
//     instead of 16x b32 for the roofline-critical 268MB adj stream
//   * packed P dword (p_even, p_odd) lands at LDS position 2*ln -> natural
//     key order, so the PV B-operand (WhcT) needs no permutation
// Per key-tile:
//   S(16x32)    = Whs(16x64) @ Wht_tile^T       -> 4x v_wmma_f32_16x16x32_f16
//   P = exp2(S) * adj  (adj in {0,1}: exact -9e15 masking; scores are O(1),
//                       softmax shift-invariance makes running-max pointless)
//   rowsum     += P @ ones(32x16)               -> 1x WMMA (every lane ends
//                                                  up holding its row's sum)
//   acc(16x128) += P(16x32) @ Whc(32x128)       -> 8x v_wmma_f32_16x16x32_f16
// Unnormalized partials (acc, row sums) go to workspace; kernel 3 combines.
// ---------------------------------------------------------------------------
__global__ __launch_bounds__(32) void gat_attn_kernel(
    const int*      __restrict__ adj,
    const _Float16* __restrict__ Whs,
    const _Float16* __restrict__ Wht,
    const _Float16* __restrict__ WhcT,
    float*          __restrict__ pacc,   // [QT*KS][16][128]
    float*          __restrict__ plsum)  // [QT*KS][16]
{
  __shared__ _Float16 pl[16][32];       // 64B rows: A-layout chunks stay b128-aligned

  const int lane = threadIdx.x & 31;
  const int ln   = lane & 15;           // N-column / M-row within tile
  const int hi   = lane >> 4;           // half-wave selector
  const int q0   = blockIdx.x << 4;     // 16 query rows per wave
  const int ks   = blockIdx.y;          // key split index
  const int jlo  = ks * (GAT_N / GAT_KS);
  const int jhik = jlo + (GAT_N / GAT_KS);

  // ---- A-operand: Whs rows q0..q0+15, K=64 as two 16x32 f16 tiles.
  // A-layout used here (16-bit, 16x32): lane L holds M=L%16;
  //   halves 0..15 -> K = 32s + 16*hi + t (contiguous, matches B)
  v16h a0, a1;
  {
    const _Float16* wr = Whs + (q0 + ln) * GAT_NH;
#pragma unroll
    for (int t = 0; t < 8; ++t) {
      a0[t]     = wr[      16 * hi + t];
      a0[8 + t] = wr[8 +  16 * hi + t];
      a1[t]     = wr[32 +  16 * hi + t];
      a1[8 + t] = wr[40 +  16 * hi + t];
    }
  }

  // Constant all-ones B tile for the row-sum WMMA.
  v16h ones;
#pragma unroll
  for (int t = 0; t < 16; ++t) ones[t] = (_Float16)1.0f;

  v8f accL = (v8f){};                   // row sums (replicated across columns)
  v8f acc[8];                           // 8 feature tiles of 16 -> 128 outputs
#pragma unroll
  for (int n = 0; n < 8; ++n) acc[n] = (v8f){};

  for (int j0 = jlo; j0 < jhik; j0 += 32) {
    // ---- scores (log2 domain): tile0 = even keys, tile1 = odd keys
    // B-layout (16-bit, 32x16): lane holds N=ln; halves 0..15 -> K = 32s + 16*hi + t
    const _Float16* wt0 = Wht + (j0 + 2 * ln) * GAT_NH + 16 * hi;  // key j0+2*ln
    const _Float16* wt1 = wt0 + GAT_NH;                            // key j0+2*ln+1
    v16h b00 = *(const v16h*)(wt0);
    v16h b01 = *(const v16h*)(wt0 + 32);
    v16h b10 = *(const v16h*)(wt1);
    v16h b11 = *(const v16h*)(wt1 + 32);

    v8f s0 = (v8f){};
    v8f s1 = (v8f){};
    s0 = __builtin_amdgcn_wmma_f32_16x16x32_f16(false, a0, false, b00, (short)0, s0, false, false);
    s0 = __builtin_amdgcn_wmma_f32_16x16x32_f16(false, a1, false, b01, (short)0, s0, false, false);
    s1 = __builtin_amdgcn_wmma_f32_16x16x32_f16(false, a0, false, b10, (short)0, s1, false, false);
    s1 = __builtin_amdgcn_wmma_f32_16x16x32_f16(false, a1, false, b11, (short)0, s1, false, false);

    // ---- P = exp2(S) * adj; adjacent (even,odd) masks -> one b64 NT load
    const int* arow = adj + (q0 + 8 * hi) * GAT_N + j0 + 2 * ln;
#pragma unroll
    for (int r = 0; r < 8; ++r) {
      const v2i ad = __builtin_nontemporal_load((const v2i*)(arow + r * GAT_N));
      union { _Float16 hh[2]; unsigned u; } pk;
      pk.hh[0] = (_Float16)(__builtin_amdgcn_exp2f(s0[r]) * (float)ad.x);
      pk.hh[1] = (_Float16)(__builtin_amdgcn_exp2f(s1[r]) * (float)ad.y);
      *(unsigned*)&pl[r + 8 * hi][2 * ln] = pk.u;   // dword-aligned packed store
    }

    __syncthreads();
    // ---- P: packed LDS -> A-layout operand; two aligned b128 reads
    const v8h* prow = (const v8h*)&pl[ln][0];       // 4x v8h per 64B row
    const v8h plo = prow[2 * hi];                   // halves: K = 16*hi + t
    const v8h phi = prow[2 * hi + 1];               // halves: K = 16*hi + 8 + t
    v16h pA = __builtin_shufflevector(plo, phi, 0, 1, 2, 3, 4, 5, 6, 7,
                                      8, 9, 10, 11, 12, 13, 14, 15);
    __syncthreads();

    // ---- row sums via WMMA against ones
    accL = __builtin_amdgcn_wmma_f32_16x16x32_f16(false, pA, false, ones, (short)0, accL, false, false);

    // ---- PV: acc(16x128) += P(16x32) @ Whc(32x128); WhcT in natural order
#pragma unroll
    for (int n = 0; n < 8; ++n) {
      const _Float16* wc = WhcT + (n * 16 + ln) * GAT_N + j0 + 16 * hi;
      v16h bv = *(const v16h*)wc;
      acc[n] = __builtin_amdgcn_wmma_f32_16x16x32_f16(false, pA, false, bv, (short)0, acc[n], false, false);
    }
  }

  // ---- epilogue: write unnormalized partials for this (qt, ks)
  const int tile = blockIdx.x * GAT_KS + ks;
  float* pa = pacc + (size_t)tile * (16 * GAT_OUTF);
#pragma unroll
  for (int r = 0; r < 8; ++r) {
    if (ln == 0) plsum[tile * 16 + r + 8 * hi] = accL[r];  // same value in all 16 lanes
#pragma unroll
    for (int n = 0; n < 8; ++n)
      pa[(r + 8 * hi) * GAT_OUTF + n * 16 + ln] = acc[n][r];
  }
}

// ---------------------------------------------------------------------------
// Kernel 3: combine split-K partials:  out = (sum_ks acc) / (sum_ks l)
// ---------------------------------------------------------------------------
__global__ __launch_bounds__(256) void gat_combine_kernel(
    const float* __restrict__ pacc,
    const float* __restrict__ plsum,
    float*       __restrict__ out)
{
  const int idx = blockIdx.x * 256 + threadIdx.x;   // q*128 + f
  const int q  = idx >> 7;
  const int f  = idx & 127;
  const int qt = q >> 4;
  const int ql = q & 15;

  float a = 0.0f, l = 0.0f;
#pragma unroll
  for (int ks = 0; ks < GAT_KS; ++ks) {
    const int t = qt * GAT_KS + ks;
    a += pacc[(size_t)t * (16 * GAT_OUTF) + ql * GAT_OUTF + f];
    l += plsum[t * 16 + ql];
  }
  out[idx] = a / l;
}

// ---------------------------------------------------------------------------
extern "C" void kernel_launch(void* const* d_in, const int* in_sizes, int n_in,
                              void* d_out, int out_size, void* d_ws, size_t ws_size,
                              hipStream_t stream) {
  const float* h   = (const float*)d_in[0];
  const int*   adj = (const int*)  d_in[1];
  const float* Ws  = (const float*)d_in[2];
  const float* Wt  = (const float*)d_in[3];
  const float* Wc  = (const float*)d_in[4];
  float*       out = (float*)d_out;

  char* ws = (char*)d_ws;
  size_t off = 0;
  _Float16* Whs  = (_Float16*)(ws + off); off += (size_t)GAT_N * GAT_NH * 2;      // 1 MB
  _Float16* Wht  = (_Float16*)(ws + off); off += (size_t)GAT_N * GAT_NH * 2;      // 1 MB
  _Float16* WhcT = (_Float16*)(ws + off); off += (size_t)GAT_N * GAT_OUTF * 2;    // 2 MB
  float* pacc  = (float*)(ws + off); off += (size_t)GAT_QT * GAT_KS * 16 * GAT_OUTF * 4; // 16 MB
  float* plsum = (float*)(ws + off);                                               // 128 KB

  hipLaunchKernelGGL(gat_proj_kernel, dim3(GAT_N), dim3(256), 0, stream,
                     h, Ws, Wt, Wc, Whs, Wht, WhcT);
  hipLaunchKernelGGL(gat_attn_kernel, dim3(GAT_QT, GAT_KS), dim3(32), 0, stream,
                     adj, Whs, Wht, WhcT, pacc, plsum);
  hipLaunchKernelGGL(gat_combine_kernel, dim3(GAT_N * GAT_OUTF / 256), dim3(256), 0, stream,
                     pacc, plsum, out);
}
